// GroupQueryAttention_15083925144391
// MI455X (gfx1250) — matrix operand order repaired and checked by reading, not compiled, not run on hardware
//
#include <hip/hip_runtime.h>
#include <hip/hip_bf16.h>

// ---------------------------------------------------------------------------
// GQA forward for B=2, S=2048, HIDDEN=2048, 32 heads (hd=64), 8 kv heads.
// All GEMMs + attention matmuls via v_wmma_f32_16x16x32_f16 (f32 accum).
// Round 2: 64x32 per-wave GEMM tile (4x2 accum) to eliminate VGPR spills.
// ---------------------------------------------------------------------------

#define HID   2048
#define NHEAD 32
#define NKV   8
#define HD    64
#define SEQ   2048
#define BATCH 2
#define MTOT  (BATCH * SEQ)        // 4096 token rows
#define NQKV  (HID + 2 * NKV * HD) // 3072

typedef __attribute__((ext_vector_type(16))) _Float16 v16h;
typedef __attribute__((ext_vector_type(8)))  _Float16 v8h;
typedef __attribute__((ext_vector_type(8)))  float    v8f;

static __device__ inline v8f wmma_f16(v16h a, v16h b, v8f c) {
  // (neg_a, A, neg_b, B, c_mod, C, reuse_a, reuse_b)
  return __builtin_amdgcn_wmma_f32_16x16x32_f16(false, a, false, b, (short)0, c,
                                                false, false);
}

// A fragment, 16x32 f16, row-major (lda elems/row).  Lane (lo,hi):
//   halves[0..15] = A[row0+lo][k0+16*hi .. +15]
static __device__ inline v16h load_afrag(const _Float16* A, size_t lda,
                                         int row0, int k0, int lo, int hi) {
  const _Float16* p = A + (size_t)(row0 + lo) * lda + (size_t)(k0 + 16 * hi);
  v8h a0 = *(const v8h*)(p);
  v8h a1 = *(const v8h*)(p + 8);
  return __builtin_shufflevector(a0, a1, 0, 1, 2, 3, 4, 5, 6, 7,
                                 8, 9, 10, 11, 12, 13, 14, 15);
}

// B fragment, 32x16 f16 from "Bt" (one row per output column, contiguous K):
//   halves[j] = B[k0+16*hi+j][col0+lo] = Bt[col0+lo][k0+16*hi+j]
static __device__ inline v16h load_bfrag(const _Float16* Bt, size_t ldb,
                                         int col0, int k0, int lo, int hi) {
  const _Float16* p = Bt + (size_t)(col0 + lo) * ldb + (size_t)(k0 + 16 * hi);
  return *(const v16h*)(p);
}

static __device__ inline v8f vzero8() {
  v8f z = {0.f, 0.f, 0.f, 0.f, 0.f, 0.f, 0.f, 0.f};
  return z;
}

// 64x32 tile per wave: 4x2 grid of 16x16 WMMA accumulators (64 VGPRs of C).
// B fragments loaded inside the N-loop to cap live registers (no spills).
static __device__ inline void gemm64x32(const _Float16* A, size_t lda,
                                        const _Float16* Bt, size_t ldb,
                                        int row0, int col0, int K,
                                        int lo, int hi, v8f acc[4][2]) {
  for (int k0 = 0; k0 < K; k0 += 32) {
    v16h a[4];
#pragma unroll
    for (int i = 0; i < 4; ++i) a[i] = load_afrag(A, lda, row0 + 16 * i, k0, lo, hi);
#pragma unroll
    for (int j = 0; j < 2; ++j) {
      const v16h b = load_bfrag(Bt, ldb, col0 + 16 * j, k0, lo, hi);
#pragma unroll
      for (int i = 0; i < 4; ++i)
        acc[i][j] = wmma_f16(a[i], b, acc[i][j]);
    }
  }
}

// ---------------------------------------------------------------------------
__global__ void f32_to_f16_kernel(const float* __restrict__ in,
                                  _Float16* __restrict__ out, int n) {
  int i = blockIdx.x * blockDim.x + threadIdx.x;
  if (i < n) out[i] = (_Float16)in[i];
}

// ---------------------------------------------------------------------------
// Fused QKV projection.  X:[4096,2048] f16, weights f16 [out,in].
// Q -> Qh[b,h,s,d], K -> Kh[b,kv,s,d], V -> Vth[b,kv,d,s] (transposed).
__global__ __launch_bounds__(32, 1)
void qkv_gemm_kernel(const _Float16* __restrict__ Xh,
                     const _Float16* __restrict__ Wqh,
                     const _Float16* __restrict__ Wkh,
                     const _Float16* __restrict__ Wvh,
                     const float* __restrict__ bq,
                     const float* __restrict__ bk,
                     const float* __restrict__ bv,
                     _Float16* __restrict__ Qh,
                     _Float16* __restrict__ Kh,
                     _Float16* __restrict__ Vth) {
  const int row0 = blockIdx.x * 64;
  const int col0 = blockIdx.y * 32;
  const int lane = threadIdx.x;
  const int lo = lane & 15, hi = lane >> 4;

  const _Float16* Bt;
  const float* bias;
  int nbase, region;  // 0=Q 1=K 2=V  (32-col tiles never straddle 2048/2560)
  if (col0 < HID)            { Bt = Wqh; bias = bq; nbase = col0;             region = 0; }
  else if (col0 < HID + 512) { Bt = Wkh; bias = bk; nbase = col0 - HID;       region = 1; }
  else                       { Bt = Wvh; bias = bv; nbase = col0 - HID - 512; region = 2; }

  v8f acc[4][2];
#pragma unroll
  for (int i = 0; i < 4; ++i)
#pragma unroll
    for (int j = 0; j < 2; ++j) acc[i][j] = vzero8();

  gemm64x32(Xh, HID, Bt, HID, row0, col0, HID, lo, hi, acc);

#pragma unroll
  for (int ni = 0; ni < 2; ++ni) {
    const int nloc = nbase + 16 * ni + lo;      // index within region
    const float bs = bias[nloc];
    const int hd = nloc >> 6, d = nloc & 63;    // head, dim within region
#pragma unroll
    for (int mi = 0; mi < 4; ++mi) {
#pragma unroll
      for (int r = 0; r < 8; ++r) {
        const int m = row0 + 16 * mi + r + 8 * hi;   // token row
        const int b = m >> 11, s = m & (SEQ - 1);
        const _Float16 h16 = (_Float16)(acc[mi][ni][r] + bs);
        if (region == 0)
          Qh[(((size_t)b * NHEAD + hd) * SEQ + s) * HD + d] = h16;
        else if (region == 1)
          Kh[(((size_t)b * NKV + hd) * SEQ + s) * HD + d] = h16;
        else
          Vth[(((size_t)b * NKV + hd) * HD + d) * SEQ + s] = h16;
      }
    }
  }
}

// ---------------------------------------------------------------------------
// Flash-style GQA attention.  One wave per (b, head, 16-query tile).
// Writes context (softmax(QK^T/8) @ V) as f16 rows of Ah[token, h*64+d].
__global__ __launch_bounds__(32, 1)
void gqa_attn_kernel(const _Float16* __restrict__ Qh,
                     const _Float16* __restrict__ Kh,
                     const _Float16* __restrict__ Vth,
                     _Float16* __restrict__ Ah) {
  const int bid = blockIdx.x;            // b*32*128 + h*128 + qt
  const int qt = bid & 127;
  const int h = (bid >> 7) & (NHEAD - 1);
  const int b = bid >> 12;
  const int q0 = qt * 16;
  const int kvh = h >> 2;                // GROUPS = 4

  const _Float16* Qhead = Qh + ((size_t)b * NHEAD + h) * SEQ * HD;
  const _Float16* Khead = Kh + ((size_t)b * NKV + kvh) * SEQ * HD;
  const _Float16* Vhead = Vth + ((size_t)b * NKV + kvh) * HD * SEQ;

  const int lane = threadIdx.x;
  const int lo = lane & 15, hi = lane >> 4;

  // Q fragments stay resident: 16 queries x 64 dims = two K-steps
  const v16h aQ0 = load_afrag(Qhead, HD, q0, 0, lo, hi);
  const v16h aQ1 = load_afrag(Qhead, HD, q0, 32, lo, hi);

  float mrow[8], lrow[8];
#pragma unroll
  for (int r = 0; r < 8; ++r) { mrow[r] = -3.0e38f; lrow[r] = 0.f; }
  v8f o[4];
#pragma unroll
  for (int dt = 0; dt < 4; ++dt) o[dt] = vzero8();

  __shared__ _Float16 P[16 * 32];  // C-layout -> A-layout transpose staging

  for (int kk = 0; kk < SEQ; kk += 32) {
    // scores for 32 keys: two 16x16 tiles, each K-dim 64 = 2 WMMAs
    v8f s0 = vzero8(), s1 = vzero8();
    s0 = wmma_f16(aQ0, load_bfrag(Khead, HD, kk, 0, lo, hi), s0);
    s0 = wmma_f16(aQ1, load_bfrag(Khead, HD, kk, 32, lo, hi), s0);
    s1 = wmma_f16(aQ0, load_bfrag(Khead, HD, kk + 16, 0, lo, hi), s1);
    s1 = wmma_f16(aQ1, load_bfrag(Khead, HD, kk + 16, 32, lo, hi), s1);

    float sc[8];
#pragma unroll
    for (int r = 0; r < 8; ++r) {
      const float v0 = s0[r] * 0.125f;     // HEAD_DIM^-0.5
      const float v1 = s1[r] * 0.125f;
      // row max across the 16 lanes holding this row (xor stays in half)
      float vmax = fmaxf(v0, v1);
      vmax = fmaxf(vmax, __shfl_xor(vmax, 1, 32));
      vmax = fmaxf(vmax, __shfl_xor(vmax, 2, 32));
      vmax = fmaxf(vmax, __shfl_xor(vmax, 4, 32));
      vmax = fmaxf(vmax, __shfl_xor(vmax, 8, 32));
      const float nm = fmaxf(mrow[r], vmax);
      sc[r] = __expf(mrow[r] - nm);
      mrow[r] = nm;
      const float p0 = __expf(v0 - nm);
      const float p1 = __expf(v1 - nm);
      float ps = p0 + p1;
      ps += __shfl_xor(ps, 1, 32);
      ps += __shfl_xor(ps, 2, 32);
      ps += __shfl_xor(ps, 4, 32);
      ps += __shfl_xor(ps, 8, 32);
      lrow[r] = lrow[r] * sc[r] + ps;
      // C-layout element (m=r+8*hi, n=lo / lo+16) -> LDS row-major 16x32
      P[(r + 8 * hi) * 32 + lo] = (_Float16)p0;
      P[(r + 8 * hi) * 32 + 16 + lo] = (_Float16)p1;
    }
    // rescale running context
#pragma unroll
    for (int dt = 0; dt < 4; ++dt)
#pragma unroll
      for (int r = 0; r < 8; ++r) o[dt][r] *= sc[r];

    __syncthreads();
    const v16h ap = load_afrag(P, 32, 0, 0, lo, hi);  // P as A 16x32
    __syncthreads();

#pragma unroll
    for (int dt = 0; dt < 4; ++dt) {
      // B = V chunk (32 keys x 16 dims) from transposed Vth: contiguous in k
      const v16h vb = load_bfrag(Vhead, SEQ, dt * 16, kk, lo, hi);
      o[dt] = wmma_f16(ap, vb, o[dt]);
    }
  }

  // normalize + store as f16 rows of [token, HID] for the output GEMM
#pragma unroll
  for (int dt = 0; dt < 4; ++dt) {
#pragma unroll
    for (int r = 0; r < 8; ++r) {
      const int m = q0 + r + 8 * hi;
      const float val = o[dt][r] / lrow[r];
      Ah[((size_t)(b * SEQ + m)) * HID + h * HD + dt * 16 + lo] = (_Float16)val;
    }
  }
}

// ---------------------------------------------------------------------------
// Output projection: out[m,n] = Ah[m,:] . Wo[n,:] + bo[n], fp32 result.
__global__ __launch_bounds__(32, 1)
void out_gemm_kernel(const _Float16* __restrict__ Ah,
                     const _Float16* __restrict__ Woh,
                     const float* __restrict__ bo,
                     float* __restrict__ out) {
  const int row0 = blockIdx.x * 64;
  const int col0 = blockIdx.y * 32;
  const int lane = threadIdx.x;
  const int lo = lane & 15, hi = lane >> 4;

  v8f acc[4][2];
#pragma unroll
  for (int i = 0; i < 4; ++i)
#pragma unroll
    for (int j = 0; j < 2; ++j) acc[i][j] = vzero8();

  gemm64x32(Ah, HID, Woh, HID, row0, col0, HID, lo, hi, acc);

#pragma unroll
  for (int ni = 0; ni < 2; ++ni) {
    const int n = col0 + 16 * ni + lo;
    const float bs = bo[n];
#pragma unroll
    for (int mi = 0; mi < 4; ++mi) {
#pragma unroll
      for (int r = 0; r < 8; ++r) {
        const int m = row0 + 16 * mi + r + 8 * hi;
        out[(size_t)m * HID + n] = acc[mi][ni][r] + bs;
      }
    }
  }
}

// ---------------------------------------------------------------------------
extern "C" void kernel_launch(void* const* d_in, const int* in_sizes, int n_in,
                              void* d_out, int out_size, void* d_ws, size_t ws_size,
                              hipStream_t stream) {
  const float* hs = (const float*)d_in[0];
  const float* Wq = (const float*)d_in[1];
  const float* bq = (const float*)d_in[2];
  const float* Wk = (const float*)d_in[3];
  const float* bk = (const float*)d_in[4];
  const float* Wv = (const float*)d_in[5];
  const float* bv = (const float*)d_in[6];
  const float* Wo = (const float*)d_in[7];
  const float* bo = (const float*)d_in[8];
  float* out = (float*)d_out;

  // f16 workspace layout (element offsets)
  _Float16* ws = (_Float16*)d_ws;
  const size_t nX = (size_t)MTOT * HID;          // 8,388,608
  const size_t nWq = (size_t)HID * HID;          // 4,194,304
  const size_t nWk = (size_t)(NKV * HD) * HID;   // 1,048,576
  _Float16* Xh  = ws;
  _Float16* Wqh = Xh + nX;
  _Float16* Wkh = Wqh + nWq;
  _Float16* Wvh = Wkh + nWk;
  _Float16* Woh = Wvh + nWk;
  _Float16* Qh  = Woh + nWq;                     // [B,32,S,64]
  _Float16* Kh  = Qh + nX;                       // [B,8,S,64]
  _Float16* Vth = Kh + (size_t)BATCH * NKV * SEQ * HD;  // [B,8,64,S]
  _Float16* Ah  = Vth + (size_t)BATCH * NKV * SEQ * HD; // [B*S, HID]

  const int cvtB = 256;
  f32_to_f16_kernel<<<(int)((nX + cvtB - 1) / cvtB), cvtB, 0, stream>>>(hs, Xh, (int)nX);
  f32_to_f16_kernel<<<(int)((nWq + cvtB - 1) / cvtB), cvtB, 0, stream>>>(Wq, Wqh, (int)nWq);
  f32_to_f16_kernel<<<(int)((nWk + cvtB - 1) / cvtB), cvtB, 0, stream>>>(Wk, Wkh, (int)nWk);
  f32_to_f16_kernel<<<(int)((nWk + cvtB - 1) / cvtB), cvtB, 0, stream>>>(Wv, Wvh, (int)nWk);
  f32_to_f16_kernel<<<(int)((nWq + cvtB - 1) / cvtB), cvtB, 0, stream>>>(Wo, Woh, (int)nWq);

  // Fused QKV projection: M=4096, N=3072, 64x32 tiles
  qkv_gemm_kernel<<<dim3(MTOT / 64, NQKV / 32), 32, 0, stream>>>(
      Xh, Wqh, Wkh, Wvh, bq, bk, bv, Qh, Kh, Vth);

  // Attention: one wave per (b, head, 16-query tile) = 2*32*128 blocks
  gqa_attn_kernel<<<BATCH * NHEAD * (SEQ / 16), 32, 0, stream>>>(Qh, Kh, Vth, Ah);

  // Output projection: M=4096, N=2048, 64x32 tiles
  out_gemm_kernel<<<dim3(MTOT / 64, HID / 32), 32, 0, stream>>>(Ah, Woh, bo, out);
}
